// STFTFusionClassifier_84018150244808
// MI455X (gfx1250) — matrix operand order repaired and checked by reading, not compiled, not run on hardware
//
#include <hip/hip_runtime.h>
#include <hip/hip_bf16.h>
#include <math.h>

typedef __attribute__((ext_vector_type(16))) _Float16 v16h;
typedef __attribute__((ext_vector_type(8)))  _Float16 v8h;
typedef __attribute__((ext_vector_type(8)))  float    v8f;

#define TNPERSEG 128
#define THOP     64
#define TNFRAMES 33
#define TNFREQ   65
#define TS       8
#define TL       2048
#define TLPAD    2176   // 2048 + 64 zero pad each side
#define KPAD1    2176   // 2145 rounded up to multiple of 32
#define NCOMP    130    // 65 re + 65 im, padded to 144 (9 tiles of 16)

static __device__ __forceinline__ v8f wmma16(v16h a, v16h b, v8f c) {
    return __builtin_amdgcn_wmma_f32_16x16x32_f16(
        /*neg_a=*/false, a, /*neg_b=*/false, b,
        /*c_mod=*/(short)0, c, /*reuse_a=*/false, /*reuse_b=*/false);
}

// ---------------------------------------------------------------------------
// softmax of the 8 sensor weights -> wsm[8] (single thread; trivial)
// ---------------------------------------------------------------------------
__global__ void softmax8_kernel(const float* __restrict__ sw, float* __restrict__ wsm) {
    if (threadIdx.x != 0 || blockIdx.x != 0) return;
    float w[8], mx = -3.0e38f;
    for (int i = 0; i < 8; ++i) { w[i] = sw[i]; mx = fmaxf(mx, w[i]); }
    float sum = 0.0f;
    for (int i = 0; i < 8; ++i) { w[i] = __expf(w[i] - mx); sum += w[i]; }
    float inv = 1.0f / sum;
    for (int i = 0; i < 8; ++i) wsm[i] = w[i] * inv;
}

// ---------------------------------------------------------------------------
// Windowed rDFT table: dft[n][k], n = component 0..143, k = sample 0..127.
// n even -> win[k]*cos(2*pi*(n/2)*k/128); n odd -> -win[k]*sin(...); pad = 0.
// N-major / K-contiguous so a B fragment is one 32B vector load.
// ---------------------------------------------------------------------------
__global__ void dft_init_kernel(_Float16* __restrict__ dft) {
    int i = blockIdx.x * blockDim.x + threadIdx.x;
    if (i >= 144 * TNPERSEG) return;
    int n = i >> 7;
    int k = i & 127;
    const float twopi = 6.2831853071795864769f;
    float win = 0.5f - 0.5f * cosf(twopi * (float)k / (float)TNPERSEG);
    float v = 0.0f;
    if (n < NCOMP) {
        int f = n >> 1;
        float ang = twopi * (float)f * (float)k / (float)TNPERSEG;
        v = (n & 1) ? (-win * sinf(ang)) : (win * cosf(ang));
    }
    dft[i] = (_Float16)v;
}

// ---------------------------------------------------------------------------
// Pre-pad + transpose + f16 convert:  x(B,L,S) f32  ->  xpad(B,S,TLPAD) f16
// 64 zeros baked in on each end so stage A needs no bounds checks.
// ---------------------------------------------------------------------------
__global__ void xpad_kernel(const float* __restrict__ x, _Float16* __restrict__ xpad,
                            int Btot) {
    long long i = (long long)blockIdx.x * blockDim.x + threadIdx.x;
    long long total = (long long)Btot * TS * TLPAD;
    if (i >= total) return;
    int l = (int)(i % TLPAD);
    long long bs = i / TLPAD;
    int s = (int)(bs % TS);
    long long b = bs / TS;
    int idx = l - (TNPERSEG / 2);
    float v = (idx >= 0 && idx < TL) ? x[(b * TL + idx) * TS + s] : 0.0f;
    xpad[i] = (_Float16)v;
}

// ---------------------------------------------------------------------------
// Pack f32 weights (Norig x Korig, row-major) into f16 (Npad x Kpad), zero-pad.
// ---------------------------------------------------------------------------
__global__ void wpack_kernel(const float* __restrict__ W, _Float16* __restrict__ Wp,
                             int Norig, int Korig, int Npad, int Kpad) {
    long long i = (long long)blockIdx.x * blockDim.x + threadIdx.x;
    long long total = (long long)Npad * Kpad;
    if (i >= total) return;
    int k = (int)(i % Kpad);
    int n = (int)(i / Kpad);
    float v = (n < Norig && k < Korig) ? W[(long long)n * Korig + k] : 0.0f;
    Wp[i] = (_Float16)v;
}

__global__ void zero_h_kernel(_Float16* __restrict__ p, int n) {
    int i = blockIdx.x * blockDim.x + threadIdx.x;
    if (i < n) p[i] = (_Float16)0.0f;
}

// ---------------------------------------------------------------------------
// Stage A: STFT magnitude + sensor fusion via WMMA.
// Virtual A matrix rows r = (b*33 + t)*8 + s, cols = sample n (0..127).
// One wave: one 16-row tile x all 9 component tiles. D-fragment VGPR r == s,
// so fusion is a register weighted sum; re/im pair via __shfl_xor(.,1).
// ---------------------------------------------------------------------------
__global__ __launch_bounds__(256)
void stft_fuse_kernel(const _Float16* __restrict__ xpad,
                      const float* __restrict__ wsm,
                      const _Float16* __restrict__ dft,
                      _Float16* __restrict__ hbuf,
                      int Btot) {
    const int lane = threadIdx.x & 31;
    const int wave = (blockIdx.x * blockDim.x + threadIdx.x) >> 5;
    const int nRowTiles = (Btot * TNFRAMES * TS) >> 4;
    if (wave >= nRowTiles) return;

    float w[8];
    #pragma unroll
    for (int i = 0; i < 8; ++i) w[i] = wsm[i];

    v8f acc[9];
    #pragma unroll
    for (int j = 0; j < 9; ++j) acc[j] = v8f{};

    const int row = (wave << 4) + (lane & 15);
    const int s   = row & 7;
    const int bt  = row >> 3;
    const int t   = bt % TNFRAMES;
    const int b   = bt / TNFRAMES;
    const _Float16* frame = xpad + ((size_t)(b * TS + s)) * TLPAD + t * THOP;
    const int ahalf = (lane < 16) ? 0 : 16;
    const int bhalf = (lane < 16) ? 0 : 16;
    const int ncol  = lane & 15;

    #pragma unroll
    for (int kc = 0; kc < 4; ++kc) {
        const int kbase = kc * 32;
        v16h a = *(const v16h*)(frame + kbase + ahalf);
        #pragma unroll
        for (int j = 0; j < 9; ++j) {
            const _Float16* bp = dft + ((size_t)(j * 16 + ncol)) * TNPERSEG
                                     + kbase + bhalf;
            v16h bf = *(const v16h*)bp;
            acc[j] = wmma16(a, bf, acc[j]);
        }
    }

    const int pairbt = wave * 2 + ((lane < 16) ? 0 : 1);
    const int pt = pairbt % TNFRAMES;
    const int pb = pairbt / TNFRAMES;
    const float inv64 = 1.0f / 64.0f;   // 1 / sum(hann)
    #pragma unroll
    for (int j = 0; j < 9; ++j) {
        int c = j * 16 + ncol;
        float fused = 0.0f;
        #pragma unroll
        for (int r = 0; r < 8; ++r) {
            float v = acc[j][r];
            float o = __shfl_xor(v, 1, 32);
            float m = sqrtf(v * v + o * o) * inv64;
            fused += w[r] * m;
        }
        if (((ncol & 1) == 0) && c < 2 * TNFREQ) {
            int f = c >> 1;
            hbuf[(size_t)pb * KPAD1 + f * TNFRAMES + pt] = (_Float16)fused;
        }
    }
}

// ---------------------------------------------------------------------------
// WMMA GEMM: out[m][n] = act( sum_k A[m][k] * Wp[n][k] + bias[n] )
// A, Wp: f16 row-major, K-padded -> pure 128-bit vector loads, no guards.
// Each wave: 1 M-tile x NTW consecutive N-tiles (A fragment reused NTW times).
// ---------------------------------------------------------------------------
template <int NTW>
__global__ __launch_bounds__(256)
void gemm_wmma_kernel(const _Float16* __restrict__ A, int lda,
                      const _Float16* __restrict__ Wp, int Kpad,
                      const float* __restrict__ bias,
                      int M, int Npad, int Norig,
                      _Float16* __restrict__ outH, float* __restrict__ outF,
                      int ldo, int relu) {
    const int lane = threadIdx.x & 31;
    const int wave = (blockIdx.x * blockDim.x + threadIdx.x) >> 5;
    const int mtiles = M >> 4;
    const int ngroups = (Npad >> 4) / NTW;
    if (wave >= mtiles * ngroups) return;
    const int mt = wave / ngroups;
    const int ng = wave % ngroups;

    const int arow = (mt << 4) + (lane & 15);
    const int asel = (lane < 16) ? 0 : 16;
    const int bsel = (lane < 16) ? 0 : 16;
    const int bn0  = ng * NTW * 16 + (lane & 15);

    v8f acc[NTW];
    #pragma unroll
    for (int j = 0; j < NTW; ++j) acc[j] = v8f{};

    const _Float16* ap0 = A + (size_t)arow * lda;
    for (int kb = 0; kb < Kpad; kb += 32) {
        v16h a = *(const v16h*)(ap0 + kb + asel);
        if (kb + 32 < Kpad)
            __builtin_prefetch((const void*)(ap0 + kb + 32), 0, 0);
        #pragma unroll
        for (int j = 0; j < NTW; ++j) {
            const _Float16* bp = Wp + (size_t)(bn0 + j * 16) * Kpad + kb + bsel;
            v16h bf = *(const v16h*)bp;
            acc[j] = wmma16(a, bf, acc[j]);
        }
    }

    #pragma unroll
    for (int j = 0; j < NTW; ++j) {
        const int ocol = ng * NTW * 16 + j * 16 + (lane & 15);
        const float bv = (ocol < Norig) ? bias[ocol] : 0.0f;
        #pragma unroll
        for (int r = 0; r < 8; ++r) {
            int orow = (mt << 4) + r + ((lane < 16) ? 0 : 8);
            float v = acc[j][r] + bv;
            if (relu) v = fmaxf(v, 0.0f);
            if (ocol < Norig) {
                if (outH) outH[(size_t)orow * ldo + ocol] = (_Float16)v;
                else      outF[(size_t)orow * ldo + ocol] = v;
            }
        }
    }
}

// ---------------------------------------------------------------------------
extern "C" void kernel_launch(void* const* d_in, const int* in_sizes, int n_in,
                              void* d_out, int out_size, void* d_ws, size_t ws_size,
                              hipStream_t stream) {
    const float* x  = (const float*)d_in[0];
    const float* sw = (const float*)d_in[1];
    const float* W1 = (const float*)d_in[2];
    const float* b1 = (const float*)d_in[3];
    const float* W2 = (const float*)d_in[4];
    const float* b2 = (const float*)d_in[5];
    const float* W3 = (const float*)d_in[6];
    const float* b3 = (const float*)d_in[7];
    const float* W4 = (const float*)d_in[8];
    const float* b4 = (const float*)d_in[9];

    const int B = in_sizes[0] / (TL * TS);   // 2048

    char* ws = (char*)d_ws;
    size_t off = 0;
    auto take = [&](size_t bytes) {
        char* p = ws + off;
        off += (bytes + 255) & ~(size_t)255;
        return p;
    };
    float*    wsm  = (float*)take(8 * sizeof(float));
    _Float16* dft  = (_Float16*)take((size_t)144 * TNPERSEG * sizeof(_Float16));
    _Float16* xpad = (_Float16*)take((size_t)B * TS * TLPAD * sizeof(_Float16));
    _Float16* h0   = (_Float16*)take((size_t)B * KPAD1 * sizeof(_Float16));
    _Float16* h1   = (_Float16*)take((size_t)B * 1024 * sizeof(_Float16));
    _Float16* h2   = (_Float16*)take((size_t)B * 512  * sizeof(_Float16));
    _Float16* h3   = (_Float16*)take((size_t)B * 256  * sizeof(_Float16));
    _Float16* w1p  = (_Float16*)take((size_t)1024 * KPAD1 * sizeof(_Float16));
    _Float16* w2p  = (_Float16*)take((size_t)512 * 1024 * sizeof(_Float16));
    _Float16* w3p  = (_Float16*)take((size_t)256 * 512 * sizeof(_Float16));
    _Float16* w4p  = (_Float16*)take((size_t)16 * 256 * sizeof(_Float16));
    (void)ws_size; (void)n_in;

    softmax8_kernel<<<1, 32, 0, stream>>>(sw, wsm);
    {
        int n = 144 * TNPERSEG;
        dft_init_kernel<<<(n + 255) / 256, 256, 0, stream>>>(dft);
    }
    {
        long long n = (long long)B * TS * TLPAD;
        xpad_kernel<<<(int)((n + 255) / 256), 256, 0, stream>>>(x, xpad, B);
    }
    {
        int n = B * KPAD1;
        zero_h_kernel<<<(n + 255) / 256, 256, 0, stream>>>(h0, n);
    }
    {   // weight packing (f32 -> f16, zero-padded K)
        long long n1 = (long long)1024 * KPAD1;
        wpack_kernel<<<(int)((n1 + 255) / 256), 256, 0, stream>>>(W1, w1p, 1024, 2145, 1024, KPAD1);
        long long n2 = (long long)512 * 1024;
        wpack_kernel<<<(int)((n2 + 255) / 256), 256, 0, stream>>>(W2, w2p, 512, 1024, 512, 1024);
        long long n3 = (long long)256 * 512;
        wpack_kernel<<<(int)((n3 + 255) / 256), 256, 0, stream>>>(W3, w3p, 256, 512, 256, 512);
        long long n4 = (long long)16 * 256;
        wpack_kernel<<<(int)((n4 + 255) / 256), 256, 0, stream>>>(W4, w4p, 10, 256, 16, 256);
    }
    {   // STFT + fusion
        int waves  = (B * TNFRAMES * TS) >> 4;     // 33792 for B=2048
        int blocks = (waves + 7) / 8;
        stft_fuse_kernel<<<blocks, 256, 0, stream>>>(xpad, wsm, dft, h0, B);
    }
    const int mtiles = B >> 4;
    {   // L1: 2145 -> 1024, relu
        int waves = mtiles * (1024 / 16 / 4);
        gemm_wmma_kernel<4><<<(waves + 7) / 8, 256, 0, stream>>>(
            h0, KPAD1, w1p, KPAD1, b1, B, 1024, 1024, h1, nullptr, 1024, 1);
    }
    {   // L2: 1024 -> 512, relu
        int waves = mtiles * (512 / 16 / 4);
        gemm_wmma_kernel<4><<<(waves + 7) / 8, 256, 0, stream>>>(
            h1, 1024, w2p, 1024, b2, B, 512, 512, h2, nullptr, 512, 1);
    }
    {   // L3: 512 -> 256, relu
        int waves = mtiles * (256 / 16 / 4);
        gemm_wmma_kernel<4><<<(waves + 7) / 8, 256, 0, stream>>>(
            h2, 512, w3p, 512, b3, B, 256, 256, h3, nullptr, 256, 1);
    }
    {   // L4: 256 -> 10, no relu, f32 out
        int waves = mtiles;
        gemm_wmma_kernel<1><<<(waves + 7) / 8, 256, 0, stream>>>(
            h3, 256, w4p, 256, b4, B, 16, 10, nullptr, (float*)d_out, 10, 0);
    }
    (void)out_size;
}
